// Decoder_38637525795297
// MI455X (gfx1250) — matrix operand order repaired and checked by reading, not compiled, not run on hardware
//
#include <hip/hip_runtime.h>
#include <hip/hip_bf16.h>

typedef __attribute__((ext_vector_type(16))) _Float16 v16h;
typedef __attribute__((ext_vector_type(8)))  _Float16 v8h;
typedef __attribute__((ext_vector_type(4)))  _Float16 v4h;
typedef __attribute__((ext_vector_type(8)))  float    v8f;
typedef __attribute__((ext_vector_type(4)))  float    v4f;

#define Bq   1024
#define Sq   256
#define Hq   128
#define Eq   128
#define G4   512
#define CDq  16
#define NT   512          // threads per decoder block (16 waves)
#define NEGQ -1.0e9f

// ---------------- native gfx1250 transcendental tanh ----------------
__device__ __forceinline__ float fast_tanh(float x) {
#if __has_builtin(__builtin_amdgcn_tanhf)
  return __builtin_amdgcn_tanhf(x);
#elif __has_builtin(__builtin_amdgcn_tanh_f32)
  return __builtin_amdgcn_tanh_f32(x);
#else
  float y;
  asm("v_tanh_f32 %0, %1" : "=v"(y) : "v"(x));
  return y;
#endif
}
__device__ __forceinline__ float fast_sigmoid(float x) {
  return 0.5f + 0.5f * fast_tanh(0.5f * x);   // 1 trans op instead of exp+rcp
}

// ---------------- WMMA helpers (gfx1250, wave32) ----------------
__device__ __forceinline__ v8f wmma_f16(v16h a, v16h b, v8f c) {
  return __builtin_amdgcn_wmma_f32_16x16x32_f16(false, a, false, b, (short)0, c,
                                                false, false);
}

// Swizzled A-fragment index for element (row m, col k), chunk-major layout:
// chunk kc = k/32; lanes 0-15 hold K {0..7,16..23}, lanes 16-31 K {8..15,24..31}.
__device__ __forceinline__ int swzA(int m, int k) {
  int kc = k >> 5, kk = k & 31;
  int lane2 = m + 16 * ((kk >> 3) & 1);
  int j = (kk & 7) + 8 * (kk >> 4);
  return (kc * 32 + lane2) * 16 + j;
}
// One contiguous 32B vector load per lane = one fragment (A or B, packed).
__device__ __forceinline__ v16h ldPk(const _Float16* pk, int tile, int lane) {
  return *reinterpret_cast<const v16h*>(pk + (((size_t)tile * 32 + lane) << 4));
}

__device__ __forceinline__ float rng01(int t, int b) {
  unsigned long long z =
      (unsigned long long)(t * Bq + b) * 0x9E3779B97F4A7C15ull + 42ull;
  z ^= z >> 30; z *= 0xBF58476D1CE4E5B9ull;
  z ^= z >> 27; z *= 0x94D049BB133111EBull;
  z ^= z >> 31;
  return (float)((z >> 40) & 0xFFFFFF) * (1.0f / 16777216.0f);
}

// ---- pack f32 W ([N][ldk]) -> f16 WMMA-B fragments (B[k][n] = W[n][k]) ----
__global__ void __launch_bounds__(32)
pack_wT(const float* __restrict__ W, int ldk, int kmax, int KC,
        _Float16* __restrict__ out) {
  int tile = blockIdx.x;                 // = ntile*KC + kc
  int kc = tile % KC, ntile = tile / KC;
  int lane = threadIdx.x;
  int n = ntile * 16 + (lane & 15), hi = lane >> 4;
  _Float16* dst = out + (((size_t)tile * 32 + lane) << 4);
#pragma unroll
  for (int j = 0; j < 16; ++j) {
    int k = kc * 32 + ((j < 8) ? (hi * 8 + j) : (16 + hi * 8 + (j - 8)));
    dst[j] = (_Float16)((k < kmax) ? W[(size_t)n * ldk + k] : 0.0f);
  }
}

// ---- pack f32 W ([M][ldk]) -> f16 WMMA-A fragments (A[m][k] = W[m][k]) ----
__global__ void __launch_bounds__(32)
pack_wA(const float* __restrict__ W, int ldk, int KC, _Float16* __restrict__ out) {
  int tile = blockIdx.x;                 // = mtile*KC + kc
  int kc = tile % KC;
  int lane = threadIdx.x;
  int m = (tile / KC) * 16 + (lane & 15), hi = lane >> 4;
  _Float16* dst = out + (((size_t)tile * 32 + lane) << 4);
#pragma unroll
  for (int j = 0; j < 8; ++j)
    dst[j]     = (_Float16)W[(size_t)m * ldk + kc * 32 + hi * 16 + j];
#pragma unroll
  for (int j = 0; j < 8; ++j)
    dst[8 + j] = (_Float16)W[(size_t)m * ldk + kc * 32 + hi * 16 + 8 + j];
}

// ---- context f32 -> f16 (same [s][b][h] layout) ----
__global__ void __launch_bounds__(512)
convert_ctx(const float* __restrict__ src, _Float16* __restrict__ dst) {
  size_t i = ((size_t)blockIdx.x * 512 + threadIdx.x) * 4;
  v4f v = *reinterpret_cast<const v4f*>(src + i);
#pragma unroll
  for (int j = 0; j < 4; ++j) dst[i + j] = (_Float16)v[j];
}

// ---- prologue: e = Wr@ctx + br, WMMA, stored f16 INTERLEAVED [b][o/8][s][8]
// so the per-step attention loop reads lane-contiguous (coalesced) v8h rows.
__global__ void __launch_bounds__(256)
precompute_e(const _Float16* __restrict__ ctx16,
             const _Float16* __restrict__ WrgA, const float* __restrict__ br_g,
             const _Float16* __restrict__ WrpA, const float* __restrict__ br_p,
             _Float16* __restrict__ eg, _Float16* __restrict__ ep) {
  int lane = threadIdx.x & 31, wave = threadIdx.x >> 5;
  int bt = blockIdx.x & 63, s = blockIdx.x >> 6;
  int b0 = bt * 16;
  int hi = lane >> 4, nl = lane & 15;

  v8f ag = {}, ap = {};
#pragma unroll
  for (int kc = 0; kc < 4; ++kc) {
    // B fragment: 16 contiguous halves straight out of ctx16 (no cvt)
    const _Float16* bp =
        ctx16 + ((size_t)s * Bq + b0 + nl) * Hq + kc * 32 + hi * 16;
    v16h bm = *reinterpret_cast<const v16h*>(bp);
    ag = wmma_f16(ldPk(WrgA, wave * 4 + kc, lane), bm, ag);
    ap = wmma_f16(ldPk(WrpA, wave * 4 + kc, lane), bm, ap);
  }
#pragma unroll
  for (int r = 0; r < 8; ++r) {
    int o = wave * 16 + 8 * hi + r;
    int b = b0 + nl;
    size_t idx = (((size_t)b * 16 + (o >> 3)) * Sq + s) * 8 + (o & 7);
    eg[idx] = (_Float16)(ag[r] + br_g[o]);
    ep[idx] = (_Float16)(ap[r] + br_p[o]);
  }
}

// ---------------- Main decoder: 64 blocks x 16 batch rows ----------------
__global__ void __launch_bounds__(NT)
decoder_main(const float* __restrict__ decoder_input,
             const float* __restrict__ embedded_inputs,
             const float* __restrict__ h0, const float* __restrict__ c0,
             const float* __restrict__ capacity,
             const unsigned char* __restrict__ attn_mask,
             const float* __restrict__ weights,
             const float* __restrict__ b_ih, const float* __restrict__ b_hh,
             const float* __restrict__ W_cap, const float* __restrict__ b_cap,
             const float* __restrict__ b_proj,
             const float* __restrict__ bq_g, const float* __restrict__ v_g,
             const float* __restrict__ bq_p, const float* __restrict__ v_p,
             const _Float16* __restrict__ Wih_pk, const _Float16* __restrict__ Whh_pk,
             const _Float16* __restrict__ Wproj_pk,
             const _Float16* __restrict__ Wqg_pk, const _Float16* __restrict__ Wqp_pk,
             const _Float16* __restrict__ eg, const _Float16* __restrict__ ep,
             const _Float16* __restrict__ ctx16,
             float* __restrict__ out_probs, float* __restrict__ out_sel) {
  const int tid = threadIdx.x, lane = tid & 31, wave = tid >> 5;
  const int b0 = blockIdx.x * 16;
  const int hi = lane >> 4, nl = lane & 15;

  // 32KB overlay: [gates 16x512 f32] OR [pq_sw 4KB | q 8KB | att 16KB]
  __shared__ __align__(32) unsigned char sBuf[16 * G4 * 4];
  float*    sG    = (float*)sBuf;
  _Float16* sPQsw = (_Float16*)sBuf;             // swizzled A buffer, 4KB
  float*    sQ    = (float*)(sBuf + 4096);       // 8KB
  float*    sAtt  = (float*)(sBuf + 12288);      // 16KB

  __shared__ float sC[16 * Hq];                          // cell state f32
  __shared__ __align__(32) _Float16 sHsw[4 * 32 * 16];   // h,  swizzled A
  __shared__ __align__(32) _Float16 sXsw[4 * 32 * 16];   // x,  swizzled A
  __shared__ __align__(32) _Float16 sAsw[5 * 32 * 16];   // [h|cap|0], swizzled A
  __shared__ _Float16 sCap[16 * CDq];
  __shared__ float sBias[G4], sBproj[Hq], sBqg[Hq], sBqp[Hq], sVg[Hq], sVp[Hq];
  __shared__ unsigned int sMask[16 * 8];    // attention_mask bit-cache
  __shared__ unsigned int sSel[16 * 8];
  __shared__ float sRem[16];
  __shared__ int   sIdx[16];

  // ---- init ----
  for (int i = tid; i < G4; i += NT) sBias[i] = b_ih[i] + b_hh[i];
  for (int i = tid; i < Hq; i += NT) {
    sBproj[i] = b_proj[i]; sBqg[i] = bq_g[i]; sBqp[i] = bq_p[i];
    sVg[i] = v_g[i]; sVp[i] = v_p[i];
  }
  for (int i = tid; i < 16 * CDq; i += NT) {
    int b = i >> 4, d = i & 15;
    sCap[i] = (_Float16)(capacity[b0 + b] * W_cap[d] + b_cap[d]);
  }
  for (int i = tid; i < 16 * Hq; i += NT) {
    int b = i >> 7, h = i & 127;
    sC[i] = c0[(size_t)(b0 + b) * Hq + h];
    sHsw[swzA(b, h)] = (_Float16)h0[(size_t)(b0 + b) * Hq + h];
    sXsw[swzA(b, h)] = (_Float16)decoder_input[(size_t)(b0 + b) * Eq + h];
  }
  for (int i = tid; i < 16 * 8; i += NT) {
    int b = i >> 3, w = i & 7;
    unsigned mw = 0;
    for (int j = 0; j < 32; ++j)
      mw |= (attn_mask[(size_t)(b0 + b) * Sq + w * 32 + j] ? 1u : 0u) << j;
    sMask[i] = mw;
    sSel[i] = 0u;
  }
  if (tid < 16) sRem[tid] = capacity[b0 + tid];
  __syncthreads();

  const v8h* egv = (const v8h*)eg;   // index: (bglob*16 + g)*Sq + s
  const v8h* epv = (const v8h*)ep;

#pragma unroll 1
  for (int t = 0; t < Sq; ++t) {
    // ---- Stage A: gates = X@W_ih^T + H@W_hh^T + bias (WMMA, 16x512) ----
    for (int wt = 0; wt < 2; ++wt) {
      int ntile = wave * 2 + wt, nbase = ntile * 16;
      v8f acc = {};
#pragma unroll
      for (int kc = 0; kc < 4; ++kc)
        acc = wmma_f16(ldPk(sXsw, kc, lane),
                       ldPk(Wih_pk, ntile * 4 + kc, lane), acc);
#pragma unroll
      for (int kc = 0; kc < 4; ++kc)
        acc = wmma_f16(ldPk(sHsw, kc, lane),
                       ldPk(Whh_pk, ntile * 4 + kc, lane), acc);
#pragma unroll
      for (int r = 0; r < 8; ++r)
        sG[(8 * hi + r) * G4 + nbase + nl] = acc[r] + sBias[nbase + nl];
    }
    __syncthreads();

    // ---- Stage B: LSTM pointwise (native v_tanh); fill concat A buffer ----
    for (int i = tid; i < 16 * Hq; i += NT) {
      int b = i >> 7, h = i & 127;
      float gi = fast_sigmoid(sG[b * G4 + h]);
      float gf = fast_sigmoid(sG[b * G4 + Hq + h]);
      float gg = fast_tanh(sG[b * G4 + 2 * Hq + h]);
      float go = fast_sigmoid(sG[b * G4 + 3 * Hq + h]);
      float c = gf * sC[i] + gi * gg;
      sC[i] = c;
      _Float16 hh = (_Float16)(go * fast_tanh(c));
      sHsw[swzA(b, h)] = hh;
      sAsw[swzA(b, h)] = hh;
      if (h < CDq)  sAsw[swzA(b, Hq + h)] = sCap[b * CDq + h];
      if (h >= 112) sAsw[swzA(b, 144 + (h - 112))] = (_Float16)0.0f;
    }
    __syncthreads();

    // ---- pq = cat@W_proj^T + bp ; q = pq@Wq^T + bq (WMMA, waves 0..7) ----
    auto proj_and_q = [&](const _Float16* Wqpk, const float* sBq) {
      if (wave < 8) {
        int nbase = wave * 16;
        v8f acc = {};
#pragma unroll
        for (int kc = 0; kc < 5; ++kc)
          acc = wmma_f16(ldPk(sAsw, kc, lane),
                         ldPk(Wproj_pk, wave * 5 + kc, lane), acc);
#pragma unroll
        for (int r = 0; r < 8; ++r) {
          int m = 8 * hi + r, n = nbase + nl;
          sPQsw[swzA(m, n)] = (_Float16)(acc[r] + sBproj[n]);
        }
      }
      __syncthreads();
      if (wave < 8) {
        int nbase = wave * 16;
        v8f acc2 = {};
#pragma unroll
        for (int kc = 0; kc < 4; ++kc)
          acc2 = wmma_f16(ldPk(sPQsw, kc, lane),
                          ldPk(Wqpk, wave * 4 + kc, lane), acc2);
#pragma unroll
        for (int r = 0; r < 8; ++r)
          sQ[(8 * hi + r) * Hq + nbase + nl] = acc2[r] + sBq[nbase + nl];
      }
      __syncthreads();
    };

    proj_and_q(Wqg_pk, sBqg);   // glimpse query qg -> sQ

    // ---- Stage D: gl = v_g . tanh(qg + e_g); coalesced v8h + v_tanh ----
    for (int p = tid; p < 16 * Sq; p += NT) {
      int b = p >> 8, s = p & 255;
      size_t base = (size_t)(b0 + b) * 16 * Sq + s;
      __builtin_prefetch((const void*)(egv + base + 32 * Sq), 0, 0);
      const v4f* q4 = (const v4f*)&sQ[b * Hq];
      float sum = 0.0f;
#pragma unroll 4
      for (int g = 0; g < 16; ++g) {
        v8h ee = egv[base + (size_t)g * Sq];
        v4f qa = q4[g * 2], qb = q4[g * 2 + 1];
#pragma unroll
        for (int j = 0; j < 4; ++j) {
          sum += sVg[g * 8 + j]     * fast_tanh(qa[j] + (float)ee[j]);
          sum += sVg[g * 8 + 4 + j] * fast_tanh(qb[j] + (float)ee[4 + j]);
        }
      }
      bool att = (sMask[b * 8 + (s >> 5)] >> (s & 31)) & 1u;
      sAtt[b * Sq + s] = att ? sum : NEGQ;
    }
    __syncthreads();

    // ---- Stage E: glimpse softmax (one wave per row) ----
    {
      int bb = wave;
      float mx = -INFINITY;
      for (int s = lane; s < Sq; s += 32) mx = fmaxf(mx, sAtt[bb * Sq + s]);
#pragma unroll
      for (int o = 16; o >= 1; o >>= 1) mx = fmaxf(mx, __shfl_xor(mx, o));
      float sm = 0.0f;
      for (int s = lane; s < Sq; s += 32) {
        float e = __expf(sAtt[bb * Sq + s] - mx);
        sAtt[bb * Sq + s] = e; sm += e;
      }
#pragma unroll
      for (int o = 16; o >= 1; o >>= 1) sm += __shfl_xor(sm, o);
      float inv = 1.0f / sm;
      for (int s = lane; s < Sq; s += 32) sAtt[bb * Sq + s] *= inv;
    }
    __syncthreads();

    // ---- Stage F: g_l[b][h] = sum_s ctx16[s][b][h]*gw[b][s]; coalesced v4h --
    {
      int b = tid >> 5, grp = tid & 31;   // one wave per row, 4 h per lane
      float a0 = 0, a1 = 0, a2 = 0, a3 = 0;
      for (int s = 0; s < Sq; ++s) {
        const v4h c4 = *(const v4h*)(ctx16 + ((size_t)s * Bq + b0 + b) * Hq + grp * 4);
        float gw = sAtt[b * Sq + s];
        a0 += (float)c4[0] * gw; a1 += (float)c4[1] * gw;
        a2 += (float)c4[2] * gw; a3 += (float)c4[3] * gw;
      }
      int h = grp * 4;
      sAsw[swzA(b, h)]     = (_Float16)a0;
      sAsw[swzA(b, h + 1)] = (_Float16)a1;
      sAsw[swzA(b, h + 2)] = (_Float16)a2;
      sAsw[swzA(b, h + 3)] = (_Float16)a3;   // cap/pad columns persist
    }
    __syncthreads();

    proj_and_q(Wqp_pk, sBqp);   // pointer query qp -> sQ

    // ---- Stage H: pointer logits + combined mask ----
    for (int p = tid; p < 16 * Sq; p += NT) {
      int b = p >> 8, s = p & 255;
      size_t base = (size_t)(b0 + b) * 16 * Sq + s;
      __builtin_prefetch((const void*)(epv + base + 32 * Sq), 0, 0);
      const v4f* q4 = (const v4f*)&sQ[b * Hq];
      float sum = 0.0f;
#pragma unroll 4
      for (int g = 0; g < 16; ++g) {
        v8h ee = epv[base + (size_t)g * Sq];
        v4f qa = q4[g * 2], qb = q4[g * 2 + 1];
#pragma unroll
        for (int j = 0; j < 4; ++j) {
          sum += sVp[g * 8 + j]     * fast_tanh(qa[j] + (float)ee[j]);
          sum += sVp[g * 8 + 4 + j] * fast_tanh(qb[j] + (float)ee[4 + j]);
        }
      }
      float logit = 10.0f * fast_tanh(sum);
      bool att  = (sMask[b * 8 + (s >> 5)] >> (s & 31)) & 1u;
      bool sel  = (sSel[b * 8 + (s >> 5)] >> (s & 31)) & 1u;
      bool comb = att && !sel && (weights[(size_t)(b0 + b) * Sq + s] <= sRem[b]);
      sAtt[b * Sq + s] = comb ? logit : NEGQ;
    }
    __syncthreads();

    // ---- Stage I: masked softmax + fallback + emit probs + parallel sample --
    {
      int bb = wave;
      float* row = &sAtt[bb * Sq];
      float* outp = out_probs + ((size_t)t * Bq + b0 + bb) * Sq;
      float mx = -INFINITY;
      for (int s = lane; s < Sq; s += 32) mx = fmaxf(mx, row[s]);
#pragma unroll
      for (int o = 16; o >= 1; o >>= 1) mx = fmaxf(mx, __shfl_xor(mx, o));
      if (mx > -1.0e8f) {
        float sm = 0.0f;
        for (int s = lane; s < Sq; s += 32) {
          float e = __expf(row[s] - mx); row[s] = e; sm += e;
        }
#pragma unroll
        for (int o = 16; o >= 1; o >>= 1) sm += __shfl_xor(sm, o);
        float inv = 1.0f / sm;
        for (int s = lane; s < Sq; s += 32) {
          float pv = row[s] * inv; row[s] = pv; outp[s] = pv;
        }
      } else {  // all-masked row: uniform over attention_mask (NaN fallback)
        float cnt = (lane < 8) ? (float)__popc(sMask[bb * 8 + lane]) : 0.0f;
#pragma unroll
        for (int o = 16; o >= 1; o >>= 1) cnt += __shfl_xor(cnt, o);
        cnt = fmaxf(cnt, 1.0f);
        float u = 1.0f / cnt;
        for (int s = lane; s < Sq; s += 32) {
          bool att = (sMask[bb * 8 + (s >> 5)] >> (s & 31)) & 1u;
          float pv = att ? u : 0.0f;
          row[s] = pv; outp[s] = pv;
        }
      }
      // wave-parallel inverse-CDF sample: chunk sums + shfl prefix + ballot
      float r = rng01(t, b0 + bb);
      float csum = 0.0f;
#pragma unroll
      for (int j = 0; j < 8; ++j) csum += row[lane * 8 + j];
      float pre = csum;
#pragma unroll
      for (int o = 1; o < 32; o <<= 1) {
        float nb = __shfl_up(pre, o);
        if (lane >= o) pre += nb;
      }
      unsigned long long bal = __ballot(pre >= r) & 0xFFFFFFFFull;
      int lsel = (bal == 0) ? 31 : (__ffsll((unsigned long long)bal) - 1);
      if (lane == lsel) {
        float c = pre - csum;
        int pick = lsel * 8 + 7;
#pragma unroll
        for (int j = 0; j < 8; ++j) {
          c += row[lane * 8 + j];
          if (c >= r) { pick = lane * 8 + j; break; }
        }
        sIdx[bb] = pick;
      }
    }
    __syncthreads();

    // ---- Stage J: state update + next decoder input (coalesced v4f) ----
    if (tid < 16) {
      int b = tid, ix = sIdx[b];
      bool att  = (sMask[b * 8 + (ix >> 5)] >> (ix & 31)) & 1u;
      bool prev = (sSel[b * 8 + (ix >> 5)] >> (ix & 31)) & 1u;
      float w = weights[(size_t)(b0 + b) * Sq + ix];
      bool valid = att && !prev && (w <= sRem[b]);
      sSel[b * 8 + (ix >> 5)] |= (1u << (ix & 31));
      if (valid) sRem[b] -= w;
      out_sel[(size_t)t * Bq + b0 + b] = (float)ix;
    }
    {
      int b = tid >> 5, grp = tid & 31;
      v4f em = *(const v4f*)&embedded_inputs[((size_t)sIdx[b] * Bq + b0 + b) * Eq + grp * 4];
      int e = grp * 4;
      sXsw[swzA(b, e)]     = (_Float16)em[0];
      sXsw[swzA(b, e + 1)] = (_Float16)em[1];
      sXsw[swzA(b, e + 2)] = (_Float16)em[2];
      sXsw[swzA(b, e + 3)] = (_Float16)em[3];
    }
    __syncthreads();
  }
}

extern "C" void kernel_launch(void* const* d_in, const int* in_sizes, int n_in,
                              void* d_out, int out_size, void* d_ws, size_t ws_size,
                              hipStream_t stream) {
  const float* decoder_input   = (const float*)d_in[0];
  const float* embedded_inputs = (const float*)d_in[1];
  const float* h0              = (const float*)d_in[2];
  const float* c0              = (const float*)d_in[3];
  const float* context         = (const float*)d_in[4];
  const float* capacity        = (const float*)d_in[5];
  const unsigned char* attn    = (const unsigned char*)d_in[6];
  const float* weights         = (const float*)d_in[7];
  const float* W_ih  = (const float*)d_in[8];
  const float* b_ih  = (const float*)d_in[9];
  const float* W_hh  = (const float*)d_in[10];
  const float* b_hh  = (const float*)d_in[11];
  const float* W_cap = (const float*)d_in[12];
  const float* b_cap = (const float*)d_in[13];
  const float* W_proj = (const float*)d_in[14];
  const float* b_proj = (const float*)d_in[15];
  const float* Wq_g = (const float*)d_in[16];
  const float* bq_g = (const float*)d_in[17];
  const float* Wr_g = (const float*)d_in[18];
  const float* br_g = (const float*)d_in[19];
  const float* v_g  = (const float*)d_in[20];
  const float* Wq_p = (const float*)d_in[21];
  const float* bq_p = (const float*)d_in[22];
  const float* Wr_p = (const float*)d_in[23];
  const float* br_p = (const float*)d_in[24];
  const float* v_p  = (const float*)d_in[25];

  const size_t NE = (size_t)Bq * Sq * Hq;  // 33.5M elements
  _Float16* eg       = (_Float16*)d_ws;            // 64MB, interleaved layout
  _Float16* ep       = eg + NE;                    // 64MB
  _Float16* ctx16    = ep + NE;                    // 64MB
  _Float16* Wih_pk   = ctx16 + NE;
  _Float16* Whh_pk   = Wih_pk + 32 * 4 * 512;
  _Float16* Wproj_pk = Whh_pk + 32 * 4 * 512;      // K=144 padded to 160
  _Float16* Wqg_pk   = Wproj_pk + 8 * 5 * 512;
  _Float16* Wqp_pk   = Wqg_pk + 8 * 4 * 512;
  _Float16* WrgA_pk  = Wqp_pk + 8 * 4 * 512;       // A-fragments for prologue
  _Float16* WrpA_pk  = WrgA_pk + 8 * 4 * 512;

  float* out_probs = (float*)d_out;                     // [S][B][S]
  float* out_sel   = out_probs + (size_t)Sq * Bq * Sq;  // [S][B]

  // one-time weight packing into f16 WMMA fragments
  pack_wT<<<dim3(32 * 4), 32, 0, stream>>>(W_ih, Eq, Eq, 4, Wih_pk);
  pack_wT<<<dim3(32 * 4), 32, 0, stream>>>(W_hh, Hq, Hq, 4, Whh_pk);
  pack_wT<<<dim3(8 * 5),  32, 0, stream>>>(W_proj, 144, 144, 5, Wproj_pk);
  pack_wT<<<dim3(8 * 4),  32, 0, stream>>>(Wq_g, Hq, Hq, 4, Wqg_pk);
  pack_wT<<<dim3(8 * 4),  32, 0, stream>>>(Wq_p, Hq, Hq, 4, Wqp_pk);
  pack_wA<<<dim3(8 * 4),  32, 0, stream>>>(Wr_g, Hq, 4, WrgA_pk);
  pack_wA<<<dim3(8 * 4),  32, 0, stream>>>(Wr_p, Hq, 4, WrpA_pk);
  // one-time context f32 -> f16
  convert_ctx<<<dim3((unsigned)(NE / (512 * 4))), 512, 0, stream>>>(context, ctx16);
  // hoisted ref projections e_g/e_p via WMMA GEMM (cvt-free, coalesced frags)
  precompute_e<<<dim3(Sq * (Bq / 16)), 256, 0, stream>>>(ctx16, WrgA_pk, br_g,
                                                         WrpA_pk, br_p, eg, ep);
  // persistent per-batch-tile decoder, full 256-step scan
  decoder_main<<<dim3(Bq / 16), NT, 0, stream>>>(
      decoder_input, embedded_inputs, h0, c0, capacity, attn, weights,
      b_ih, b_hh, W_cap, b_cap, b_proj, bq_g, v_g, bq_p, v_p,
      Wih_pk, Whh_pk, Wproj_pk, Wqg_pk, Wqp_pk, eg, ep, ctx16,
      out_probs, out_sel);
}
